// Attention_26852135534981
// MI455X (gfx1250) — matrix operand order repaired and checked by reading, not compiled, not run on hardware
//
#include <hip/hip_runtime.h>

// ---------------------------------------------------------------------------
// Fused attention block for MI455X (gfx1250, wave32, WMMA).
//   x[4,2048,1024] f32 -> LN -> QKV gemm -> 16-head flash attention -> out proj
// All matmuls use v_wmma_f32_16x16x32_f16 (f16 in, f32 accumulate).
// Data movement: GEMM A tiles via global_load_async_to_lds_b128 (ASYNCcnt),
// attention K tiles via the Tensor Data Mover (tensor_load_to_lds, TENSORcnt),
// transposed tiles (B^T, V^T) via manual LDS scatter.
// ---------------------------------------------------------------------------

#define TOKENS  8192
#define SEQ     2048
#define BATCH   4
#define DIMM    1024
#define NHEADS  16
#define DHEAD   64
#define QKVW    3072          // 3 * HEADS * DHEAD
#define SCALE_F 0.125f        // DHEAD^-0.5

typedef __attribute__((ext_vector_type(16))) _Float16 v16h;
typedef __attribute__((ext_vector_type(8)))  _Float16 v8h;
typedef __attribute__((ext_vector_type(8)))  float    v8f;
typedef __attribute__((ext_vector_type(4)))  unsigned int u32x4;
typedef __attribute__((ext_vector_type(8)))  int          i32x8;
typedef __attribute__((ext_vector_type(4)))  int          i32x4;

static __device__ __forceinline__ v8h ld8(const _Float16* p) {
    return *(const v8h*)p;
}
static __device__ __forceinline__ v16h cat16(v8h lo, v8h hi) {
    return __builtin_shufflevector(lo, hi, 0,1,2,3,4,5,6,7,8,9,10,11,12,13,14,15);
}
static __device__ __forceinline__ v8f wmma_f16(v16h a, v16h b, v8f c) {
    // 8 args: (neg_a, A, neg_b, B, c_mod, C, reuse_a, reuse_b)
    return __builtin_amdgcn_wmma_f32_16x16x32_f16(false, a, false, b,
                                                  (short)0, c, false, false);
}
static __device__ __forceinline__ v8f zero8f() {
    v8f z;
#pragma unroll
    for (int i = 0; i < 8; ++i) z[i] = 0.0f;
    return z;
}

// LDS aperture occupies addr[63:32] only (ISA 10.2), so the low 32 bits of a
// generic pointer to a __shared__ object ARE the LDS byte address.
static __device__ __forceinline__ unsigned lds_addr_of(const void* p) {
    return (unsigned)(unsigned long long)p;
}

// Per-lane async copy of 2x16B global -> LDS (GV mode). The 24-bit instruction
// offset is added to BOTH the LDS and the global address (ISA 10.x async
// pseudocode), so one base pair serves both 16B halves.
static __device__ __forceinline__ void async_copy32B(unsigned lds,
                                                     const void* gaddr) {
    unsigned long long g = (unsigned long long)gaddr;
    asm volatile("global_load_async_to_lds_b128 %0, %1, off"
                 :: "v"(lds), "v"(g) : "memory");
    asm volatile("global_load_async_to_lds_b128 %0, %1, off offset:16"
                 :: "v"(lds), "v"(g) : "memory");
}
static __device__ __forceinline__ void wait_async() {
    asm volatile("s_wait_asynccnt 0x0" ::: "memory");
}

// TDM: one instruction DMAs a whole 2D tile (tile_w x tile_h f16 elements,
// row stride in elements) from global to LDS. D# packed per ISA 8.3/8.4:
// group0 = {count=1, lds_addr, global_addr[56:0], type=2}, group1 holds
// data_size=2B, tensor dims, tile dims, dim0 stride. Groups 2/3 unused (2D).
// amdgpu-toolchain (clang-23) builtin arity: 6 args, trailing i32x8 + cpol.
static __device__ __forceinline__ void tdm_load_2d(unsigned lds,
                                                   const void* gptr,
                                                   unsigned tile_w,
                                                   unsigned tile_h,
                                                   unsigned stride_elems) {
    unsigned long long ga = (unsigned long long)gptr;
    u32x4 g0;
    g0[0] = 1u;                                   // count=1, user descriptor
    g0[1] = lds;                                  // lds_addr (bytes)
    g0[2] = (unsigned)ga;                         // global_addr[31:0]
    g0[3] = (unsigned)((ga >> 32) & 0x01FFFFFFu)  // global_addr[56:32]
            | (2u << 30);                         // type = 2 ("image")
    i32x8 g1;
    g1[0] = (int)(1u << 16);                      // data_size = 1 -> 2 bytes
    g1[1] = (int)((tile_w & 0xFFFFu) << 16);      // tensor_dim0[15:0]
    g1[2] = (int)(((tile_w >> 16) & 0xFFFFu)      // tensor_dim0[31:16]
            | ((tile_h & 0xFFFFu) << 16));        // tensor_dim1[15:0]
    g1[3] = (int)(((tile_h >> 16) & 0xFFFFu)      // tensor_dim1[31:16]
            | ((tile_w & 0xFFFFu) << 16));        // tile_dim0
    g1[4] = (int)(tile_h & 0xFFFFu);              // tile_dim1 (tile_dim2=0)
    g1[5] = (int)stride_elems;                    // tensor_dim0_stride[31:0]
    g1[6] = 0;                                    // stride hi / dim1_stride
    g1[7] = 0;
    i32x4 z4 = {0, 0, 0, 0};
    i32x8 z8;
#pragma unroll
    for (int i = 0; i < 8; ++i) z8[i] = 0;
    __builtin_amdgcn_tensor_load_to_lds(g0, g1, z4, z4, z8, 0);
}

// ---------------------------------------------------------------------------
// f32 -> f16 elementwise convert (weights)
// ---------------------------------------------------------------------------
__global__ void cvt_f32_f16(const float* __restrict__ in,
                            _Float16* __restrict__ out, int n) {
    int i = blockIdx.x * blockDim.x + threadIdx.x;
    int stride = gridDim.x * blockDim.x;
    for (; i < n; i += stride) out[i] = (_Float16)in[i];
}

// ---------------------------------------------------------------------------
// LayerNorm over dim=1024, one token per 256-thread block, f16 output.
// ---------------------------------------------------------------------------
__global__ __launch_bounds__(256)
void ln_kernel(const float* __restrict__ x, const float* __restrict__ gamma,
               const float* __restrict__ beta, _Float16* __restrict__ xn) {
    __shared__ float ssum[256];
    __shared__ float ssq[256];
    const int t = threadIdx.x;
    const size_t base = (size_t)blockIdx.x * DIMM;

    float4 v = *(const float4*)(x + base + t * 4);
    ssum[t] = v.x + v.y + v.z + v.w;
    ssq[t]  = v.x * v.x + v.y * v.y + v.z * v.z + v.w * v.w;
    __syncthreads();
#pragma unroll
    for (int s = 128; s > 0; s >>= 1) {
        if (t < s) { ssum[t] += ssum[t + s]; ssq[t] += ssq[t + s]; }
        __syncthreads();
    }
    const float mean = ssum[0] * (1.0f / DIMM);
    const float var  = ssq[0] * (1.0f / DIMM) - mean * mean;
    const float inv  = rsqrtf(var + 1e-5f);

    const float4 g = *(const float4*)(gamma + t * 4);
    const float4 b = *(const float4*)(beta + t * 4);
    xn[base + t * 4 + 0] = (_Float16)((v.x - mean) * inv * g.x + b.x);
    xn[base + t * 4 + 1] = (_Float16)((v.y - mean) * inv * g.y + b.y);
    xn[base + t * 4 + 2] = (_Float16)((v.z - mean) * inv * g.z + b.z);
    xn[base + t * 4 + 3] = (_Float16)((v.w - mean) * inv * g.w + b.w);
}

// ---------------------------------------------------------------------------
// WMMA GEMM: C[M,N] = A[M,K] * B[K,N], A/B f16 row-major.
// Block = 128 threads (4 waves), 64x64 tile; wave w -> rows [w*16, w*16+16).
// A tile staged via async LDS loads; B^T via manual scatter (transpose).
// ---------------------------------------------------------------------------
template <bool F32OUT>
__global__ __launch_bounds__(128)
void gemm16(const _Float16* __restrict__ A, const _Float16* __restrict__ B,
            void* __restrict__ Cp, int M, int N, int K) {
    __shared__ __align__(16) _Float16 sA[64 * 32];
    __shared__ __align__(16) _Float16 sBT[64 * 32];

    const int tid  = threadIdx.x;
    const int lane = tid & 31;
    const int wave = tid >> 5;
    const int lrow = lane & 15;
    const int hi   = lane >> 4;
    const int n0 = blockIdx.x * 64;
    const int m0 = blockIdx.y * 64;
    const unsigned sA_lds = lds_addr_of(&sA[0]);

    v8f acc[4];
#pragma unroll
    for (int s = 0; s < 4; ++s) acc[s] = zero8f();

    for (int k0 = 0; k0 < K; k0 += 32) {
        __syncthreads();
        {   // stage A 64x32 asynchronously (row-major, stride 32)
            const int row = tid >> 1, c = (tid & 1) * 16;
            async_copy32B(sA_lds + (unsigned)(row * 32 + c) * 2,
                          A + (size_t)(m0 + row) * K + k0 + c);
            if (k0 + 32 < K)
                __builtin_prefetch(A + (size_t)(m0 + row) * K + k0 + 32 + c, 0, 1);
        }
        {   // stage B^T 64x32: sBT[n][k] = B[k0+k][n0+n]
            const int kk = tid >> 2, c = (tid & 3) * 16;
            const _Float16* g = B + (size_t)(k0 + kk) * N + n0 + c;
            v8h b0 = ld8(g), b1 = ld8(g + 8);
#pragma unroll
            for (int i = 0; i < 8; ++i) {
                sBT[(c + i) * 32 + kk]     = b0[i];
                sBT[(c + 8 + i) * 32 + kk] = b1[i];
            }
            if (k0 + 32 < K)
                __builtin_prefetch(B + (size_t)(k0 + 32 + kk) * N + n0 + c, 0, 1);
        }
        wait_async();
        __syncthreads();

        const int aoff = hi ? 8 : 0;
        const v16h afrag = cat16(ld8(&sA[(wave * 16 + lrow) * 32 + aoff]),
                                 ld8(&sA[(wave * 16 + lrow) * 32 + aoff + 16]));
        const int boff = hi ? 8 : 0;
#pragma unroll
        for (int s = 0; s < 4; ++s) {
            const v16h bfrag = cat16(ld8(&sBT[(s * 16 + lrow) * 32 + boff]),
                                     ld8(&sBT[(s * 16 + lrow) * 32 + boff + 16]));
            acc[s] = wmma_f16(afrag, bfrag, acc[s]);
        }
    }

    // C/D layout: lane holds (row = r + 8*hi, col = lrow) for VGPR r in [0,8)
#pragma unroll
    for (int s = 0; s < 4; ++s) {
        const int col = n0 + s * 16 + lrow;
#pragma unroll
        for (int r = 0; r < 8; ++r) {
            const size_t row = (size_t)(m0 + wave * 16 + r + 8 * hi);
            if (F32OUT) ((float*)Cp)[row * N + col] = acc[s][r];
            else        ((_Float16*)Cp)[row * N + col] = (_Float16)acc[s][r];
        }
    }
}

// ---------------------------------------------------------------------------
// Flash attention. qkv: [tokens, 3072] f16 (q | k | v, each head-major n x 64).
// Block = 4 waves sharing (batch, head); wave handles 16 query rows.
// K tile (32x64) staged by the Tensor Data Mover; V^T via manual scatter.
// ---------------------------------------------------------------------------
__global__ __launch_bounds__(128)
void attn_kernel(const _Float16* __restrict__ qkv, _Float16* __restrict__ aout) {
    __shared__ __align__(16) _Float16 sK[32 * 64];
    __shared__ __align__(16) _Float16 sVT[64 * 32];
    __shared__ __align__(16) _Float16 pbuf[4][16 * 32];

    const int tid  = threadIdx.x;
    const int lane = tid & 31;
    const int wave = tid >> 5;
    const int lrow = lane & 15;
    const int hi   = lane >> 4;
    const int b = blockIdx.z;
    const int h = blockIdx.y;
    const int i0 = blockIdx.x * 64 + wave * 16;
    const size_t tok0 = (size_t)b * SEQ;
    const unsigned sK_lds = lds_addr_of(&sK[0]);

    // Q fragments: 2 K-steps of 32 over d=64, held in registers for all keys
    v16h qf[2];
    {
        const size_t rowq = (tok0 + i0 + lrow) * QKVW + h * DHEAD;
#pragma unroll
        for (int t = 0; t < 2; ++t) {
            const int off = t * 32 + (hi ? 8 : 0);
            qf[t] = cat16(ld8(qkv + rowq + off), ld8(qkv + rowq + off + 16));
        }
    }

    float m[8], l[8];
    v8f o[4];
#pragma unroll
    for (int r = 0; r < 8; ++r) { m[r] = -1e30f; l[r] = 0.0f; }
#pragma unroll
    for (int s = 0; s < 4; ++s) o[s] = zero8f();

    const int kcol = DIMM + h * DHEAD;
    const int vcol = 2 * DIMM + h * DHEAD;

    for (int jt = 0; jt < SEQ; jt += 32) {
        __syncthreads();   // protect LDS tiles from previous iteration's readers
        {   // stage V transposed (manual scatter, all threads)
            const int kk = tid >> 2, c = (tid & 3) * 16;
            const _Float16* gv = qkv + (tok0 + jt + kk) * QKVW + vcol + c;
            v8h v0 = ld8(gv), v1 = ld8(gv + 8);
#pragma unroll
            for (int i = 0; i < 8; ++i) {
                sVT[(c + i) * 32 + kk]     = v0[i];
                sVT[(c + 8 + i) * 32 + kk] = v1[i];
            }
            if (jt + 32 < SEQ) {
                __builtin_prefetch(qkv + (tok0 + jt + 32 + kk) * QKVW + vcol + c, 0, 1);
                __builtin_prefetch(qkv + (tok0 + jt + 32 + kk) * QKVW + kcol + c, 0, 1);
            }
        }
        // stage K 32x64 row-major with one TDM descriptor (wave 0 issues)
        if (wave == 0) {
            tdm_load_2d(sK_lds, qkv + (tok0 + jt) * QKVW + kcol,
                        /*tile_w=*/DHEAD, /*tile_h=*/32, /*stride=*/QKVW);
            __builtin_amdgcn_s_wait_tensorcnt(0);
        }
        __syncthreads();

        // S = Q * K^T for 32 keys -> two 16x16 f32 tiles
        v8f s0 = zero8f(), s1 = zero8f();
#pragma unroll
        for (int t = 0; t < 2; ++t) {
            const int doff = t * 32 + (hi ? 8 : 0);
            const v16h bk0 = cat16(ld8(&sK[lrow * 64 + doff]),
                                   ld8(&sK[lrow * 64 + doff + 16]));
            s0 = wmma_f16(qf[t], bk0, s0);
            const v16h bk1 = cat16(ld8(&sK[(16 + lrow) * 64 + doff]),
                                   ld8(&sK[(16 + lrow) * 64 + doff + 16]));
            s1 = wmma_f16(qf[t], bk1, s1);
        }

        // streaming softmax; row r+8*hi lives across the 16 lanes of this half
        _Float16 ph0[8], ph1[8];
#pragma unroll
        for (int r = 0; r < 8; ++r) {
            const float a0 = s0[r] * SCALE_F;
            const float a1 = s1[r] * SCALE_F;
            float mx = fmaxf(a0, a1);
#pragma unroll
            for (int off = 1; off < 16; off <<= 1)
                mx = fmaxf(mx, __shfl_xor(mx, off, 16));
            const float mn    = fmaxf(m[r], mx);
            const float alpha = __expf(m[r] - mn);
            const float p0 = __expf(a0 - mn);
            const float p1 = __expf(a1 - mn);
            float rs = p0 + p1;
#pragma unroll
            for (int off = 1; off < 16; off <<= 1)
                rs += __shfl_xor(rs, off, 16);
            l[r] = l[r] * alpha + rs;
            m[r] = mn;
#pragma unroll
            for (int s = 0; s < 4; ++s) o[s][r] *= alpha;
            ph0[r] = (_Float16)p0;
            ph1[r] = (_Float16)p1;
        }

        // P: C-layout -> A-layout via per-wave LDS scratch (16x32 f16)
#pragma unroll
        for (int r = 0; r < 8; ++r) {
            const int row = r + 8 * hi;
            pbuf[wave][row * 32 + lrow]      = ph0[r];
            pbuf[wave][row * 32 + 16 + lrow] = ph1[r];
        }
        __syncthreads();

        const int poff = hi ? 8 : 0;
        const v16h pf = cat16(ld8(&pbuf[wave][lrow * 32 + poff]),
                              ld8(&pbuf[wave][lrow * 32 + poff + 16]));
        const int voff = hi ? 8 : 0;
#pragma unroll
        for (int s = 0; s < 4; ++s) {
            const v16h vf = cat16(ld8(&sVT[(s * 16 + lrow) * 32 + voff]),
                                  ld8(&sVT[(s * 16 + lrow) * 32 + voff + 16]));
            o[s] = wmma_f16(pf, vf, o[s]);
        }
    }

    // normalize by l and write attn_out [token, 1024] f16
#pragma unroll
    for (int r = 0; r < 8; ++r) {
        const float inv = 1.0f / l[r];
        const size_t row = tok0 + i0 + r + 8 * hi;
#pragma unroll
        for (int s = 0; s < 4; ++s)
            aout[row * DIMM + h * DHEAD + s * 16 + lrow] =
                (_Float16)(o[s][r] * inv);
    }
}

// ---------------------------------------------------------------------------
// Host entry. Inputs: x, ln_gamma, ln_beta, w_qkv, w_out (all f32).
// ---------------------------------------------------------------------------
extern "C" void kernel_launch(void* const* d_in, const int* in_sizes, int n_in,
                              void* d_out, int out_size, void* d_ws, size_t ws_size,
                              hipStream_t stream) {
    (void)in_sizes; (void)n_in; (void)out_size; (void)ws_size;

    const float* x    = (const float*)d_in[0];
    const float* gam  = (const float*)d_in[1];
    const float* bet  = (const float*)d_in[2];
    const float* wqkv = (const float*)d_in[3];
    const float* wout = (const float*)d_in[4];
    float* out = (float*)d_out;

    // workspace carve-up (f16 buffers, ~88 MB total)
    char* p = (char*)d_ws;
    _Float16* xn    = (_Float16*)p; p += (size_t)TOKENS * DIMM * 2;
    _Float16* wqkvh = (_Float16*)p; p += (size_t)DIMM * QKVW * 2;
    _Float16* wouth = (_Float16*)p; p += (size_t)DIMM * DIMM * 2;
    _Float16* qkvb  = (_Float16*)p; p += (size_t)TOKENS * QKVW * 2;
    _Float16* aout  = (_Float16*)p; p += (size_t)TOKENS * DIMM * 2;

    cvt_f32_f16<<<2048, 256, 0, stream>>>(wqkv, wqkvh, DIMM * QKVW);
    cvt_f32_f16<<<1024, 256, 0, stream>>>(wout, wouth, DIMM * DIMM);
    ln_kernel<<<TOKENS, 256, 0, stream>>>(x, gam, bet, xn);

    // QKV projection: [8192,1024] x [1024,3072] -> f16
    gemm16<false><<<dim3(QKVW / 64, TOKENS / 64), 128, 0, stream>>>(
        xn, wqkvh, qkvb, TOKENS, QKVW, DIMM);

    // flash attention
    attn_kernel<<<dim3(SEQ / 64, NHEADS, BATCH), 128, 0, stream>>>(qkvb, aout);

    // output projection: [8192,1024] x [1024,1024] -> f32 into d_out
    gemm16<true><<<dim3(DIMM / 64, TOKENS / 64), 128, 0, stream>>>(
        aout, wouth, out, TOKENS, DIMM, DIMM);
}
